// BaseModelWithEmbedding_83176336654766
// MI455X (gfx1250) — matrix operand order repaired and checked by reading, not compiled, not run on hardware
//
#include <hip/hip_runtime.h>
#include <hip/hip_bf16.h>

typedef __attribute__((ext_vector_type(16))) _Float16 v16h;
typedef __attribute__((ext_vector_type(8)))  _Float16 v8h;
typedef __attribute__((ext_vector_type(8)))  float    v8f;

#define B_TOT 256
#define T_LEN 512
#define H_DIM 128
#define G4    512   // 4*H
#define BB    32    // batch rows per block
#define CHUNK 64    // timesteps prefetched per LDS chunk

// single-instruction v_rcp_f32 instead of the precise-division sequence
__device__ __forceinline__ float rcp_(float x) { return __builtin_amdgcn_rcpf(x); }
__device__ __forceinline__ float sigmoidf_(float x) {
    return rcp_(1.0f + __expf(-x));
}
__device__ __forceinline__ float tanhf_(float x) {
    x = fminf(fmaxf(x, -15.0f), 15.0f);
    float e = __expf(2.0f * x);
    return fmaf(-2.0f, rcp_(e + 1.0f), 1.0f);   // 1 - 2/(e^{2x}+1)
}

// ---------------------------------------------------------------------------
// Kernel 0: precompute xz gather tables:
//   tabs[0 .. 24*512)        = time_emb @ W_h + b_h   (per hour index)
//   tabs[24*512 .. 31*512)   = week_emb @ W_w + b_w   (per weekday index)
// ---------------------------------------------------------------------------
__global__ __launch_bounds__(256) void xz_table_kernel(
    const float* __restrict__ time_emb, const float* __restrict__ week_emb,
    const float* __restrict__ W_h, const float* __restrict__ b_h,
    const float* __restrict__ W_w, const float* __restrict__ b_w,
    float* __restrict__ tabs)
{
    int i = blockIdx.x * 256 + threadIdx.x;
    const int NH = 24 * G4;
    const int NW = 7 * G4;
    if (i < NH) {
        int r = i >> 9, c = i & (G4 - 1);
        float acc = b_h[c];
        #pragma unroll 4
        for (int k = 0; k < H_DIM; ++k)
            acc = fmaf(time_emb[r * H_DIM + k], W_h[k * G4 + c], acc);
        tabs[i] = acc;
    } else if (i < NH + NW) {
        int j = i - NH;
        int r = j >> 9, c = j & (G4 - 1);
        float acc = b_w[c];
        #pragma unroll 4
        for (int k = 0; k < H_DIM; ++k)
            acc = fmaf(week_emb[r * H_DIM + k], W_w[k * G4 + c], acc);
        tabs[i] = acc;
    }
}

// ---------------------------------------------------------------------------
// Kernel 1: fused LSTM scan.  grid.x = 3 LSTMs * 8 batch-chunks of 32 rows.
// 8 waves; wave w owns gate columns g*128 + [16w,16w+16) for g=i,f,g,o so the
// gate combine and cell state stay wave-local.  Recurrent matmul per step:
// h[32,128](LDS,f16) @ U[128,512](registers) via v_wmma_f32_16x16x32_f16.
// ---------------------------------------------------------------------------
__global__ __launch_bounds__(256, 1) void lstm_scan_kernel(
    const float* __restrict__ spatial,   // [B,T,2]
    const int*   __restrict__ hour_idx,  // [B,T]
    const int*   __restrict__ week_idx,  // [B,T]
    const float* __restrict__ W_sp,      // [2,512]
    const float* __restrict__ b_sp,      // [512]
    const float* __restrict__ U_sp,      // [128,512]
    const float* __restrict__ U_h,
    const float* __restrict__ U_w,
    const float* __restrict__ tabs,      // [24*512 | 7*512]
    float* __restrict__ hout)            // [3,256,128]
{
    __shared__ alignas(16) _Float16 h_sh[2][BB][H_DIM];   // 16 KB, double buffered
    __shared__ alignas(16) _Float16 tab_sh[24 * G4];      // 24 KB (f16 xz table / W_sp+b_sp)
    __shared__ alignas(16) int      chunk_sh[CHUNK * BB * 2]; // 16 KB idx or spatial chunk

    const int tid     = threadIdx.x;
    const int lane    = tid & 31;          // wave32
    const int wv      = tid >> 5;          // 8 waves
    const int lane_lo = lane & 15;
    const int lanehi  = lane >> 4;
    const int lstm    = blockIdx.x >> 3;   // 0=spatial 1=hour 2=week
    const int b0      = (blockIdx.x & 7) * BB;
    const int ncol    = wv * 16 + lane_lo; // this lane's hidden column (0..127)

    const float* U = (lstm == 0) ? U_sp : (lstm == 1) ? U_h : U_w;

    // ---- stage the xz table (or spatial weights) into LDS as f16 ----
    if (lstm == 0) {
        for (int i = tid; i < 2 * G4; i += 256) tab_sh[i] = (_Float16)W_sp[i];
        for (int i = tid; i < G4; i += 256)     tab_sh[2 * G4 + i] = (_Float16)b_sp[i];
    } else {
        const float* tab = (lstm == 1) ? tabs : (tabs + 24 * G4);
        const int n = (lstm == 1) ? 24 * G4 : 7 * G4;
        for (int i = tid; i < n; i += 256)      tab_sh[i] = (_Float16)tab[i];
    }
    // h(t=0) = 0
    for (int i = tid; i < BB * H_DIM; i += 256)
        (&h_sh[0][0][0])[i] = (_Float16)0.0f;

    // ---- load U as WMMA B-operands (K=32 x N=16 tiles), kept in VGPRs ----
    // B layout: lane -> column n = lane%16; lane half selects K+8;
    // element e packs k_local = lanehi*8 + e (+8 for e >= 8).
    v16h Breg[4][4];
    #pragma unroll
    for (int g = 0; g < 4; ++g) {
        #pragma unroll
        for (int kc = 0; kc < 4; ++kc) {
            v16h bv;
            const int col = g * H_DIM + ncol;
            #pragma unroll
            for (int e = 0; e < 16; ++e) {
                const int k = kc * 32 + lanehi * 8 + e + (e >= 8 ? 8 : 0);
                bv[e] = (_Float16)U[k * G4 + col];
            }
            Breg[g][kc] = bv;
        }
    }

    // cell state in registers: element (m,n): m = bt*16 + lanehi*8 + j, n = ncol
    float cst[2][8];
    #pragma unroll
    for (int bt = 0; bt < 2; ++bt)
        #pragma unroll
        for (int j = 0; j < 8; ++j) cst[bt][j] = 0.0f;

    __syncthreads();

    int buf = 0;
    for (int t = 0; t < T_LEN; ++t) {
        const int tl = t & (CHUNK - 1);
        if (tl == 0) {            // refill per-step input chunk (coalesced)
            __syncthreads();
            if (lstm == 0) {
                for (int i = tid; i < BB * CHUNK * 2; i += 256) {
                    const int b = i / (CHUNK * 2);
                    const int rem = i - b * (CHUNK * 2);
                    const int u = rem >> 1, c2 = rem & 1;
                    chunk_sh[(u * BB + b) * 2 + c2] =
                        __float_as_int(spatial[((size_t)(b0 + b) * T_LEN + t + u) * 2 + c2]);
                }
            } else {
                const int* idxp = (lstm == 1) ? hour_idx : week_idx;
                for (int i = tid; i < BB * CHUNK; i += 256) {
                    const int b = i / CHUNK;
                    const int u = i - b * CHUNK;
                    chunk_sh[u * BB + b] = idxp[(b0 + b) * T_LEN + t + u];
                }
            }
            __syncthreads();
        }
        const int nbuf = buf ^ 1;

        #pragma unroll
        for (int bt = 0; bt < 2; ++bt) {
            const int m_base = bt * 16 + lanehi * 8;
            v8f acc[4];
            // ---- init accumulators with the input contribution xz_t ----
            #pragma unroll
            for (int j = 0; j < 8; ++j) {
                const int bl = m_base + j;
                if (lstm == 0) {
                    const float s0 = __int_as_float(chunk_sh[(tl * BB + bl) * 2 + 0]);
                    const float s1 = __int_as_float(chunk_sh[(tl * BB + bl) * 2 + 1]);
                    #pragma unroll
                    for (int g = 0; g < 4; ++g) {
                        const int col = g * H_DIM + ncol;
                        acc[g][j] = fmaf(s0, (float)tab_sh[col],
                                    fmaf(s1, (float)tab_sh[G4 + col],
                                         (float)tab_sh[2 * G4 + col]));
                    }
                } else {
                    const int idx = chunk_sh[tl * BB + bl];
                    #pragma unroll
                    for (int g = 0; g < 4; ++g)
                        acc[g][j] = (float)tab_sh[idx * G4 + g * H_DIM + ncol];
                }
            }
            // ---- preload all 4 A-tiles so DS latency overlaps the WMMAs ----
            union AU { v16h v; v8h h8[2]; } au[4];
            #pragma unroll
            for (int kc = 0; kc < 4; ++kc) {
                // A layout (16-bit 16x32): lane m = lane%16; half selects K+8;
                // elems 0..7 = K[kc*32 + lanehi*8 ..], 8..15 = +16.
                const _Float16* base = &h_sh[buf][bt * 16 + lane_lo][kc * 32 + lanehi * 8];
                au[kc].h8[0] = *(const v8h*)base;
                au[kc].h8[1] = *(const v8h*)(base + 16);
            }
            // ---- z += h_{t-1} @ U : 16 WMMAs, A tile shared by 4 gates ----
            #pragma unroll
            for (int kc = 0; kc < 4; ++kc) {
                #pragma unroll
                for (int g = 0; g < 4; ++g)
                    acc[g] = __builtin_amdgcn_wmma_f32_16x16x32_f16(
                        false, au[kc].v, false, Breg[g][kc], (short)0, acc[g], false, false);
            }
            // ---- gates, cell update, write h_t ----
            #pragma unroll
            for (int j = 0; j < 8; ++j) {
                const float i_ = sigmoidf_(acc[0][j]);
                const float f_ = sigmoidf_(acc[1][j]);
                const float g_ = tanhf_(acc[2][j]);
                const float o_ = sigmoidf_(acc[3][j]);
                const float c  = fmaf(f_, cst[bt][j], i_ * g_);
                cst[bt][j] = c;
                const float h = o_ * tanhf_(c);
                const int m = m_base + j;
                h_sh[nbuf][m][ncol] = (_Float16)h;
                if (t == T_LEN - 1)
                    hout[((size_t)lstm * B_TOT + b0 + m) * H_DIM + ncol] = h;
            }
        }
        __syncthreads();
        buf = nbuf;
    }
}

// ---------------------------------------------------------------------------
// Kernel 2: out[b] = concat(h_sp,h_h,h_w)[b] . fc_W + fc_b  (tiny)
// ---------------------------------------------------------------------------
__global__ __launch_bounds__(256) void fc_kernel(
    const float* __restrict__ hout, const float* __restrict__ fc_W,
    const float* __restrict__ fc_b, float* __restrict__ out)
{
    const int b = blockIdx.x * 256 + threadIdx.x;
    if (b < B_TOT) {
        float acc = fc_b[0];
        #pragma unroll
        for (int l = 0; l < 3; ++l)
            for (int n = 0; n < H_DIM; ++n)
                acc = fmaf(hout[((size_t)l * B_TOT + b) * H_DIM + n],
                           fc_W[l * H_DIM + n], acc);
        out[b] = acc;
    }
}

extern "C" void kernel_launch(void* const* d_in, const int* in_sizes, int n_in,
                              void* d_out, int out_size, void* d_ws, size_t ws_size,
                              hipStream_t stream) {
    const float* spatial  = (const float*)d_in[0];
    const int*   hour_idx = (const int*)  d_in[1];
    const int*   week_idx = (const int*)  d_in[2];
    const float* time_emb = (const float*)d_in[3];
    const float* week_emb = (const float*)d_in[4];
    const float* W_sp = (const float*)d_in[5];
    const float* U_sp = (const float*)d_in[6];
    const float* b_sp = (const float*)d_in[7];
    const float* W_h  = (const float*)d_in[8];
    const float* U_h  = (const float*)d_in[9];
    const float* b_h  = (const float*)d_in[10];
    const float* W_w  = (const float*)d_in[11];
    const float* U_w  = (const float*)d_in[12];
    const float* b_w  = (const float*)d_in[13];
    const float* fc_W = (const float*)d_in[14];
    const float* fc_b = (const float*)d_in[15];

    float* ws   = (float*)d_ws;
    float* tabs = ws;                 // 31*512 floats
    float* hout = ws + 31 * G4;       // 3*256*128 floats

    xz_table_kernel<<<62, 256, 0, stream>>>(time_emb, week_emb, W_h, b_h, W_w, b_w, tabs);
    lstm_scan_kernel<<<24, 256, 0, stream>>>(spatial, hour_idx, week_idx,
                                             W_sp, b_sp, U_sp, U_h, U_w, tabs, hout);
    fc_kernel<<<1, 256, 0, stream>>>(hout, fc_W, fc_b, (float*)d_out);
}
